// StructuredAttention_23450521436609
// MI455X (gfx1250) — matrix operand order repaired and checked by reading, not compiled, not run on hardware
//
#include <hip/hip_runtime.h>

// Problem constants (fixed by the reference)
#define BB 8
#define NN 512
#define DD 768
#define HH 12
#define DK 64
#define NEGV (-1e9f)
// final score = 0.9 * raw/sqrt(64) + 0.1*dep = 0.1125*raw + 0.1*dep
#define SCL 0.1125f
#define DWT 0.1f
// padded LDS row stride: 516 mod 64 = 4 -> conflict-free WMMA dump
#define NP (NN + 4)

typedef __attribute__((ext_vector_type(16))) __bf16 v16bf;
typedef __attribute__((ext_vector_type(8)))  __bf16 v8bf;
typedef __attribute__((ext_vector_type(8)))  float  v8f;
typedef __attribute__((ext_vector_type(4)))  float  v4f;

// round-to-nearest-even f32 -> bf16 (bit pattern as ushort)
__device__ __forceinline__ unsigned short f2bf(float f) {
  union { float f; unsigned u; } v; v.f = f;
  unsigned u = v.u;
  return (unsigned short)((u + 0x7FFFu + ((u >> 16) & 1u)) >> 16);
}

// ---------------------------------------------------------------------------
// Kernel 1: f32 -> bf16 bulk convert (vectorized float4 -> ushort4)
// ---------------------------------------------------------------------------
__global__ void k_convert(const float* __restrict__ in,
                          unsigned short* __restrict__ out, int n4) {
  int i = blockIdx.x * blockDim.x + threadIdx.x;
  if (i >= n4) return;
  v4f v = ((const v4f*)in)[i];
  ushort4 o;
  o.x = f2bf(v.x); o.y = f2bf(v.y); o.z = f2bf(v.z); o.w = f2bf(v.w);
  ((ushort4*)out)[i] = o;
}

// ---------------------------------------------------------------------------
// Kernel 2: transpose + convert weight: WT[n][k] = bf16(W[k][n])
// ---------------------------------------------------------------------------
__global__ void k_transpose_bf(const float* __restrict__ w,
                               unsigned short* __restrict__ wt) {
  int id = blockIdx.x * blockDim.x + threadIdx.x;
  if (id >= DD * DD) return;
  int n = id / DD, k = id % DD;
  wt[id] = f2bf(w[k * DD + n]);   // wt[n*DD + k]
}

// ---------------------------------------------------------------------------
// Kernel 3: projection GEMM  out[m][n] = bf16( sum_k X[m][k]*W[k][n] + b[n] )
// Software-pipelined: A/B fragments double-buffered so loads for step i+1
// are in flight while WMMA i executes (avoids s_wait_loadcnt 0 per WMMA).
// ---------------------------------------------------------------------------
__global__ __launch_bounds__(128) void k_proj(
    const unsigned short* __restrict__ X,
    const unsigned short* __restrict__ WT,
    const float* __restrict__ bias,
    unsigned short* __restrict__ out) {
  const int lane = threadIdx.x & 31;
  const int wv   = threadIdx.x >> 5;
  const int m0   = blockIdx.x * 16;
  const int nb   = blockIdx.y * 16 + wv * 4;  // first of 4 ntiles for this wave

  // A-fragment lane mapping (16-bit A 16x32): lanes 0-15 -> K 0..15,
  // lanes 16-31 -> K 16..31, row M = lane&15.
  const int mrow  = lane & 15;
  const int khalf = (lane >> 4) * 16;
  // B-fragment lane mapping (32x16): col N = lane&15; lanes 0-15 K 0..15,
  // lanes 16-31 K 16..31 (contiguous 32B in WT row).
  const int ncol = lane & 15;
  const int kq   = (lane >> 4) * 16;

  const unsigned short* paBase = X + (m0 + mrow) * DD;

  auto loadA = [&](int kofs) -> v16bf {
    v8bf lo = *(const v8bf*)(paBase + kofs + khalf);
    v8bf hi = *(const v8bf*)(paBase + kofs + 8 + khalf);
    v16bf a;
#pragma unroll
    for (int i = 0; i < 8; ++i) { a[i] = lo[i]; a[8 + i] = hi[i]; }
    return a;
  };
  auto loadB = [&](int kofs, int t) -> v16bf {
    const int n = (nb + t) * 16 + ncol;
    return *(const v16bf*)(WT + n * DD + kofs + kq);
  };

  v8f c[4];
#pragma unroll
  for (int t = 0; t < 4; ++t) c[t] = (v8f){};

  v16bf a_cur = loadA(0);
  v16bf b_cur = loadB(0, 0);

  for (int ks = 0; ks < 24; ++ks) {
    const int kofs = ks * 32;
    v16bf a_nxt = a_cur;
    if (ks + 1 < 24) a_nxt = loadA(kofs + 32);
#pragma unroll
    for (int t = 0; t < 4; ++t) {
      v16bf b_nxt = b_cur;
      if (t < 3)            b_nxt = loadB(kofs, t + 1);
      else if (ks + 1 < 24) b_nxt = loadB(kofs + 32, 0);
      c[t] = __builtin_amdgcn_wmma_f32_16x16x32_bf16(
          false, a_cur, false, b_cur, (short)0, c[t], false, false);
      b_cur = b_nxt;
    }
    a_cur = a_nxt;
  }

  // C/D layout: vgpr r, lanes 0-15 -> M=r, lanes 16-31 -> M=8+r; N = lane&15.
#pragma unroll
  for (int t = 0; t < 4; ++t) {
    const int n = (nb + t) * 16 + ncol;
    const float bv = bias[n];
#pragma unroll
    for (int r = 0; r < 8; ++r) {
      const int row = m0 + r + (lane >> 4) * 8;
      out[row * DD + n] = f2bf(c[t][r] + bv);
    }
  }
}

// ---------------------------------------------------------------------------
// Kernel 4: fused scores + blend + mask + softmax, single HBM write of p_attn.
// grid = (N/16, H, B), block = 128 (4 waves). 16 query rows x 512 key cols
// per workgroup; scores staged in padded LDS. B fragments double-buffered.
// ---------------------------------------------------------------------------
__global__ __launch_bounds__(128) void k_attn(
    const unsigned short* __restrict__ Q,   // [B*N, 768] bf16
    const unsigned short* __restrict__ K,   // [B*N, 768] bf16
    const float* __restrict__ dep,          // [B, N, N]
    const int* __restrict__ mask,           // [B, 1, N]
    float* __restrict__ out) {              // [B, H, N, N]
  __shared__ float s[16][NP];
  __shared__ float red[16][8];

  const int b  = blockIdx.z;
  const int h  = blockIdx.y;
  const int m0 = blockIdx.x * 16;
  const int lane = threadIdx.x & 31;
  const int wv   = threadIdx.x >> 5;

  const int mrow  = lane & 15;
  const int khalf = (lane >> 4) * 16;
  const int ncol  = lane & 15;
  const int kq    = (lane >> 4) * 16;

  const int row = threadIdx.x >> 3;   // 0..15 (softmax phase assignment)
  const int seg = threadIdx.x & 7;    // 0..7
  const int mg  = m0 + row;
  const float* drow = dep + (b * NN + mg) * NN;

  // Warm L2 with this block's dep_emb segment while WMMAs run (256B/thread)
  __builtin_prefetch(drow + seg * 64, 0, 1);
  __builtin_prefetch(drow + seg * 64 + 32, 0, 1);

  // A fragments: Q rows m0..m0+15, head slice h*64..h*64+63 (two k=32 frags)
  const unsigned short* pq = Q + ((b * NN) + m0 + mrow) * DD + h * DK;
  v16bf a0, a1;
  {
    v8bf l0 = *(const v8bf*)(pq + khalf);
    v8bf h0 = *(const v8bf*)(pq + 8 + khalf);
    v8bf l1 = *(const v8bf*)(pq + 32 + khalf);
    v8bf h1 = *(const v8bf*)(pq + 40 + khalf);
#pragma unroll
    for (int i = 0; i < 8; ++i) {
      a0[i] = l0[i]; a0[8 + i] = h0[i];
      a1[i] = l1[i]; a1[8 + i] = h1[i];
    }
  }

  auto loadBpair = [&](int nt, v16bf& b0, v16bf& b1) {
    const int n0 = (wv * 8 + nt) * 16;
    // B operand for Q.K^T: element (k,n) = Kmat[n][k] -> contiguous in K row.
    const unsigned short* pk = K + ((b * NN) + n0 + ncol) * DD + h * DK;
    b0 = *(const v16bf*)(pk + kq);
    b1 = *(const v16bf*)(pk + 32 + kq);
  };

  // Each wave: 8 column tiles of 16 -> whole 512-wide row block covered.
  v16bf b0, b1;
  loadBpair(0, b0, b1);
  for (int nt = 0; nt < 8; ++nt) {
    v16bf nb0 = b0, nb1 = b1;
    if (nt + 1 < 8) loadBpair(nt + 1, nb0, nb1);   // prefetch next tile
    v8f c = (v8f){};
    c = __builtin_amdgcn_wmma_f32_16x16x32_bf16(false, a0, false, b0,
                                                (short)0, c, false, false);
    c = __builtin_amdgcn_wmma_f32_16x16x32_bf16(false, a1, false, b1,
                                                (short)0, c, false, false);
    const int n0 = (wv * 8 + nt) * 16;
#pragma unroll
    for (int r = 0; r < 8; ++r)
      s[r + (lane >> 4) * 8][n0 + ncol] = c[r];
    b0 = nb0; b1 = nb1;
  }
  __syncthreads();

  // Cooperative softmax: 8 threads per row, 64 cols each.
  const int* mk = mask + b * NN;

  // Pass 1: blend + mask, track max
  float mx = -3.4e38f;
#pragma unroll 4
  for (int j = 0; j < 16; ++j) {
    const int col = seg * 64 + j * 4;
    v4f d = *(const v4f*)(drow + col);
    int4 mm = *(const int4*)(mk + col);
#pragma unroll
    for (int i = 0; i < 4; ++i) {
      const int mbit = (i == 0) ? mm.x : (i == 1) ? mm.y : (i == 2) ? mm.z : mm.w;
      float v = SCL * s[row][col + i] + DWT * d[i];
      if (mbit == 0) v = NEGV;
      s[row][col + i] = v;
      mx = fmaxf(mx, v);
    }
  }
  red[row][seg] = mx;
  __syncthreads();
#pragma unroll
  for (int i = 0; i < 8; ++i) mx = fmaxf(mx, red[row][i]);

  // Pass 2: exp + partial sums
  float sum = 0.f;
  for (int j = 0; j < 64; ++j) {
    const int col = seg * 64 + j;
    float e = __expf(s[row][col] - mx);
    s[row][col] = e;
    sum += e;
  }
  __syncthreads();            // everyone done reading red (max) before reuse
  red[row][seg] = sum;
  __syncthreads();
  float tot = 0.f;
#pragma unroll
  for (int i = 0; i < 8; ++i) tot += red[row][i];
  const float inv = 1.0f / tot;

  // Pass 3: single vectorized HBM write of probabilities
  float* orow = out + (((size_t)(b * HH + h) * NN) + mg) * NN;
#pragma unroll 4
  for (int j = 0; j < 16; ++j) {
    const int col = seg * 64 + j * 4;
    v4f o;
    o.x = s[row][col + 0] * inv;
    o.y = s[row][col + 1] * inv;
    o.z = s[row][col + 2] * inv;
    o.w = s[row][col + 3] * inv;
    *(v4f*)(orow + col) = o;
  }
}

// ---------------------------------------------------------------------------
extern "C" void kernel_launch(void* const* d_in, const int* in_sizes, int n_in,
                              void* d_out, int out_size, void* d_ws, size_t ws_size,
                              hipStream_t stream) {
  const float* x    = (const float*)d_in[0];  // [B,N,D]
  const int*   msk  = (const int*)d_in[1];    // [B,1,N]
  const float* dep  = (const float*)d_in[2];  // [B,N,N]
  const float* Wq   = (const float*)d_in[3];  // [D,D]
  const float* bq   = (const float*)d_in[4];  // [D]
  const float* Wk   = (const float*)d_in[5];  // [D,D]
  const float* bk   = (const float*)d_in[6];  // [D]
  float* out = (float*)d_out;

  // Workspace layout (bf16 as ushort): ~20.3 MB total
  unsigned short* Xbf = (unsigned short*)d_ws;            // 4096*768
  unsigned short* WqT = Xbf + (size_t)BB * NN * DD;       // 768*768
  unsigned short* WkT = WqT + (size_t)DD * DD;            // 768*768
  unsigned short* Qbf = WkT + (size_t)DD * DD;            // 4096*768
  unsigned short* Kbf = Qbf + (size_t)BB * NN * DD;       // 4096*768

  const int nX4 = BB * NN * DD / 4;
  k_convert<<<(nX4 + 255) / 256, 256, 0, stream>>>(x, Xbf, nX4);
  k_transpose_bf<<<(DD * DD + 255) / 256, 256, 0, stream>>>(Wq, WqT);
  k_transpose_bf<<<(DD * DD + 255) / 256, 256, 0, stream>>>(Wk, WkT);

  // Q and K projections: 256 row-tiles x 3 groups of 16 col-tiles
  k_proj<<<dim3(256, 3, 1), 128, 0, stream>>>(Xbf, WqT, bq, Qbf);
  k_proj<<<dim3(256, 3, 1), 128, 0, stream>>>(Xbf, WkT, bk, Kbf);

  // Fused attention probabilities
  k_attn<<<dim3(NN / 16, HH, BB), 128, 0, stream>>>(Qbf, Kbf, dep, msk, out);
}
